// WaveNet_91018946937510
// MI455X (gfx1250) — matrix operand order repaired and checked by reading, compiled, not run on hardware
//
#include <hip/hip_runtime.h>
#include <hip/hip_bf16.h>

typedef __attribute__((ext_vector_type(16))) _Float16 v16h;
typedef __attribute__((ext_vector_type(8)))  _Float16 v8h;
typedef __attribute__((ext_vector_type(8)))  float    v8f;

#define B_   4
#define L_   10239
#define BL   40956        // B_*L_
#define BLP  40960        // padded to 32-col tiles
#define NTP  1280         // BLP/32 column-pair tiles

__device__ __forceinline__ v8f wmma16(v16h a, v16h b, v8f c) {
    return __builtin_amdgcn_wmma_f32_16x16x32_f16(false, a, false, b, (short)0, c, false, false);
}

// A-fragment (16x32 f16) from LDS, W row-major [M][ldk]
__device__ __forceinline__ v16h load_A(const _Float16* W, int ldk, int m0, int k0, int lane) {
    const _Float16* p = W + (size_t)(m0 + (lane & 15)) * ldk + k0 + ((lane & 16) ? 16 : 0);
    v8h lo = *(const v8h*)p;
    v8h hi = *(const v8h*)(p + 8);
    v16h a;
#pragma unroll
    for (int i = 0; i < 8; i++) { a[i] = lo[i]; a[i + 8] = hi[i]; }
    return a;
}

// B-fragment (32x16 f16) from global, act t-major [t][ldc]; columns N = t0..t0+15
__device__ __forceinline__ v16h load_B(const _Float16* act, int ldc, int t0, int k0, int lane) {
    const _Float16* p = act + (size_t)(t0 + (lane & 15)) * ldc + k0 + ((lane & 16) ? 16 : 0);
    return *(const v16h*)p;
}

// Stage weights global->LDS via async-to-LDS DMA (ASYNCcnt path, no VGPR round-trip).
// The LDS byte address is derived from the actual __shared__ pointer (low 32 bits of
// the generic address = LDS offset); the ptrtoint also makes the LDS object escape so
// the asm "memory" clobber is known to write it (keeps the later ds_loads alive).
// 256 threads x 16B = 4096B per iteration.
__device__ __forceinline__ void stage_lds_async(void* lds0, const void* g0, unsigned nbytes) {
    const char* g = (const char*)g0 + threadIdx.x * 16;
    unsigned lds = (unsigned)(uintptr_t)lds0 + threadIdx.x * 16u;
    unsigned iters = nbytes >> 12;
    for (unsigned i = 0; i < iters; i++) {
        asm volatile("global_load_async_to_lds_b128 %0, %1, off"
                     :: "v"(lds), "v"(g) : "memory");
        lds += 4096u;
        g += 4096;
    }
    asm volatile("s_wait_asynccnt 0x0" ::: "memory");
}

// ---------------- prep kernels (fp32 weights -> f16 fused layouts) ----------------

__global__ void k_prep_w1(const float* __restrict__ cond_w, const float* __restrict__ dil_w,
                          _Float16* __restrict__ W1) {
    int idx = blockIdx.x * blockDim.x + threadIdx.x;   // 16*256*320
    int blk = idx / 81920, rem = idx % 81920;
    int r = rem / 320, k = rem % 320;
    int rr = r & 127;
    float v = 0.f;
    if (rr < 120) {
        int orow = (r < 128) ? rr : (120 + rr);
        if (k < 80)       v = cond_w[(blk * 240 + orow) * 80 + k];
        else if (k < 200) v = dil_w[((blk * 240 + orow) * 120 + (k - 80)) * 2 + 0];
        else              v = dil_w[((blk * 240 + orow) * 120 + (k - 200)) * 2 + 1];
    }
    W1[idx] = (_Float16)v;
}

__global__ void k_prep_gbias(const float* __restrict__ cond_b, const float* __restrict__ dil_b,
                             float* __restrict__ gbias) {
    int idx = blockIdx.x * blockDim.x + threadIdx.x;   // 16*256
    int blk = idx / 256, r = idx % 256, rr = r & 127;
    float v = 0.f;
    if (rr < 120) {
        int orow = (r < 128) ? rr : (120 + rr);
        v = cond_b[blk * 240 + orow] + dil_b[blk * 240 + orow];
    }
    gbias[idx] = v;
}

__global__ void k_prep_a2(const float* __restrict__ skip_w, const float* __restrict__ res_w,
                          _Float16* __restrict__ A2) {
    int idx = blockIdx.x * blockDim.x + threadIdx.x;   // 16*368*128
    int blk = idx / (368 * 128), rem = idx % (368 * 128);
    int r = rem / 128, k = rem % 128;
    float v = 0.f;
    if (k < 120) {
        if (r < 240)      v = skip_w[(blk * 240 + r) * 120 + k];
        else if (r < 360) v = res_w[(blk * 120 + (r - 240)) * 120 + k];
    }
    A2[idx] = (_Float16)v;
}

__global__ void k_prep_bias2(const float* __restrict__ skip_b, const float* __restrict__ res_b,
                             float* __restrict__ bias2) {
    int idx = blockIdx.x * blockDim.x + threadIdx.x;   // 16*368
    int blk = idx / 368, r = idx % 368;
    float v = 0.f;
    if (r < 240)      v = skip_b[blk * 240 + r];
    else if (r < 360) v = res_b[blk * 120 + (r - 240)];
    bias2[idx] = v;
}

__global__ void k_prep_outw(const float* __restrict__ out_w, _Float16* __restrict__ W) {
    int idx = blockIdx.x * blockDim.x + threadIdx.x;   // 256*256
    int o = idx / 256, k = idx % 256;
    W[idx] = (_Float16)((k < 240) ? out_w[o * 240 + k] : 0.f);
}

__global__ void k_prep_endw(const float* __restrict__ end_w, _Float16* __restrict__ W) {
    int idx = blockIdx.x * blockDim.x + threadIdx.x;   // 256*256
    W[idx] = (_Float16)end_w[idx];
}

// ---------------- frontend: transposed-conv upsample + causal conv ----------------

__global__ void k_upsample(const float* __restrict__ mel, const float* __restrict__ up_w,
                           const float* __restrict__ up_b, _Float16* __restrict__ mel_up) {
    int idx = blockIdx.x * blockDim.x + threadIdx.x;   // BLP*80
    int t = idx / 80, o = idx % 80;
    float acc = 0.f;
    if (t < BL) {
        int b = t / L_, l = t - b * L_;
        int m1 = (l + 257) >> 8;
        int kk1 = (l + 257) - (m1 << 8);   // in [0,255]
        acc = up_b[o];
        const float* melb = mel + b * 80 * 41;
        if (m1 <= 40) {
#pragma unroll 4
            for (int i = 0; i < 80; i++) acc += melb[i * 41 + m1] * up_w[(i * 80 + o) * 512 + kk1];
        }
        if (m1 >= 1) {
            int kk0 = kk1 + 256;
#pragma unroll 4
            for (int i = 0; i < 80; i++) acc += melb[i * 41 + (m1 - 1)] * up_w[(i * 80 + o) * 512 + kk0];
        }
    }
    mel_up[(size_t)t * 80 + o] = (_Float16)acc;
}

__global__ void k_causal(const float* __restrict__ wav, const float* __restrict__ causal_w,
                         const float* __restrict__ causal_b, _Float16* __restrict__ h) {
    int idx = blockIdx.x * blockDim.x + threadIdx.x;   // BLP*128
    int t = idx >> 7, r = idx & 127;
    float acc = 0.f;
    if (t < BL && r < 120) {
        int b = t / L_, l = t - b * L_;
        acc = causal_b[r];
        const float* wv = wav + (size_t)b * L_;
        const float* cw = causal_w + r * 256;
        int k0 = 255 - l; if (k0 < 0) k0 = 0;
        for (int k = k0; k < 256; k++) acc += cw[k] * wv[l + k - 255];
    }
    h[(size_t)t * 128 + r] = (_Float16)acc;
}

__global__ void k_zero(float* __restrict__ p, size_t n) {
    size_t i = (size_t)blockIdx.x * blockDim.x + threadIdx.x;
    size_t stride = (size_t)gridDim.x * blockDim.x;
    for (; i < n; i += stride) p[i] = 0.f;
}

// pack per-position K=320 input vector: [mel_up(80) | h[t-d](120) | h[t](120)]
__global__ void k_pack(const _Float16* __restrict__ mel_up, const _Float16* __restrict__ h,
                       _Float16* __restrict__ packed, int d) {
    int idx = blockIdx.x * blockDim.x + threadIdx.x;   // BLP*320
    int t = idx / 320, j = idx % 320;
    _Float16 v = (_Float16)0.f;
    if (t < BL) {
        if (j < 80) v = mel_up[(size_t)t * 80 + j];
        else if (j < 200) {
            int l = t % L_;
            if (l >= d) v = h[(size_t)(t - d) * 128 + (j - 80)];
        } else v = h[(size_t)t * 128 + (j - 200)];
    }
    packed[(size_t)t * 320 + j] = v;
}

// ---------------- WMMA GEMM kernels (32 columns per wave: 2 C-tiles share A frags) ----

// gate: g(256 rows) = W1(256x320) @ packed ; xg = tanh(gT)*sigmoid(gS), f16 [t][128]
__global__ void __launch_bounds__(256) k_gemm_gate(const _Float16* __restrict__ packed,
                                                   const _Float16* __restrict__ W1,
                                                   const float* __restrict__ gbias,
                                                   _Float16* __restrict__ xg) {
    __shared__ _Float16 sW[256 * 320];                 // 160 KB
    stage_lds_async(sW, W1, 256 * 320 * 2);
    __syncthreads();
    int wave = threadIdx.x >> 5, lane = threadIdx.x & 31;
    int tile = blockIdx.x * 8 + wave;
    if (tile >= NTP) return;
    int t0 = tile * 32;
    int t  = t0 + (lane & 15);
    int mb = (lane & 16) ? 8 : 0;

    v16h bf0[10], bf1[10];
    {
        const _Float16* p0 = packed + (size_t)t * 320 + ((lane & 16) ? 16 : 0);
        const _Float16* p1 = p0 + 16 * 320;
#pragma unroll
        for (int k = 0; k < 10; k++) { bf0[k] = *(const v16h*)(p0 + k * 32);
                                       bf1[k] = *(const v16h*)(p1 + k * 32); }
    }

#pragma unroll
    for (int p = 0; p < 8; p++) {
        v8f accT0 = {}, accT1 = {}, accS0 = {}, accS1 = {};
#pragma unroll
        for (int k = 0; k < 10; k++) {
            v16h a = load_A(sW, 320, p * 16, k * 32, lane);
            accT0 = wmma16(a, bf0[k], accT0);
            accT1 = wmma16(a, bf1[k], accT1);
        }
#pragma unroll
        for (int k = 0; k < 10; k++) {
            v16h a = load_A(sW, 320, 128 + p * 16, k * 32, lane);
            accS0 = wmma16(a, bf0[k], accS0);
            accS1 = wmma16(a, bf1[k], accS1);
        }
        v8h o0, o1;
#pragma unroll
        for (int r = 0; r < 8; r++) {
            float bT = gbias[p * 16 + r + mb];
            float bS = gbias[128 + p * 16 + r + mb];
            o0[r] = (_Float16)(tanhf(accT0[r] + bT) * (1.0f / (1.0f + __expf(-(accS0[r] + bS)))));
            o1[r] = (_Float16)(tanhf(accT1[r] + bT) * (1.0f / (1.0f + __expf(-(accS1[r] + bS)))));
        }
        *(v8h*)(xg + (size_t)t * 128 + p * 16 + mb)        = o0;  // pad rows 120..127 -> 0
        *(v8h*)(xg + (size_t)(t + 16) * 128 + p * 16 + mb) = o1;
    }
}

// skip/res: [skip(240); res(120)] = A2(368x128) @ xg ; skips += , h += res
__global__ void __launch_bounds__(256) k_gemm_sr(const _Float16* __restrict__ xg,
                                                 const _Float16* __restrict__ A2,
                                                 const float* __restrict__ bias2,
                                                 float* __restrict__ skips,
                                                 _Float16* __restrict__ h) {
    __shared__ _Float16 sW[368 * 128];                 // 94 KB
    stage_lds_async(sW, A2, 368 * 128 * 2);
    __syncthreads();
    int wave = threadIdx.x >> 5, lane = threadIdx.x & 31;
    int tile = blockIdx.x * 8 + wave;
    if (tile >= NTP) return;
    int t0 = tile * 32;
    int t  = t0 + (lane & 15);
    int mb = (lane & 16) ? 8 : 0;

    v16h bf0[4], bf1[4];
#pragma unroll
    for (int k = 0; k < 4; k++) { bf0[k] = load_B(xg, 128, t0,      k * 32, lane);
                                  bf1[k] = load_B(xg, 128, t0 + 16, k * 32, lane); }

#pragma unroll
    for (int mt = 0; mt < 23; mt++) {
        v8f acc0 = {}, acc1 = {};
#pragma unroll
        for (int k = 0; k < 4; k++) {
            v16h a = load_A(sW, 128, mt * 16, k * 32, lane);
            acc0 = wmma16(a, bf0[k], acc0);
            acc1 = wmma16(a, bf1[k], acc1);
        }
        int rb = mt * 16 + mb;
        if (mt < 15) {
            float* sp0 = skips + (size_t)t * 256 + rb;
            float* sp1 = sp0 + 16 * 256;
#pragma unroll
            for (int r = 0; r < 8; r++) {
                float bb = bias2[rb + r];
                sp0[r] += acc0[r] + bb;
                sp1[r] += acc1[r] + bb;
            }
        } else {
#pragma unroll
            for (int r = 0; r < 8; r++) {
                int c = rb + r - 240;
                if (c < 120) {
                    float bb = bias2[rb + r];
                    size_t hi0 = (size_t)t * 128 + c;
                    size_t hi1 = hi0 + 16 * 128;
                    h[hi0] = (_Float16)((float)h[hi0] + acc0[r] + bb);
                    h[hi1] = (_Float16)((float)h[hi1] + acc1[r] + bb);
                }
            }
        }
    }
}

// out head: act2 = relu(out_w @ relu(skips) + out_b), f16 [t][256]
__global__ void __launch_bounds__(256) k_gemm_out(const float* __restrict__ skips,
                                                  const _Float16* __restrict__ outW,
                                                  const float* __restrict__ out_b,
                                                  _Float16* __restrict__ act2) {
    __shared__ _Float16 sW[256 * 256];                 // 128 KB
    stage_lds_async(sW, outW, 256 * 256 * 2);
    __syncthreads();
    int wave = threadIdx.x >> 5, lane = threadIdx.x & 31;
    int tile = blockIdx.x * 8 + wave;
    if (tile >= NTP) return;
    int t0 = tile * 32;
    int t  = t0 + (lane & 15);
    int mb = (lane & 16) ? 8 : 0;

    v16h bf0[8], bf1[8];
#pragma unroll
    for (int k = 0; k < 8; k++) {
        const float* p0 = skips + (size_t)t * 256 + k * 32 + ((lane & 16) ? 16 : 0);
        const float* p1 = p0 + 16 * 256;
        v16h b0, b1;
#pragma unroll
        for (int i = 0; i < 16; i++) { b0[i] = (_Float16)fmaxf(p0[i], 0.f);
                                       b1[i] = (_Float16)fmaxf(p1[i], 0.f); }
        bf0[k] = b0; bf1[k] = b1;
    }
#pragma unroll
    for (int mt = 0; mt < 16; mt++) {
        v8f acc0 = {}, acc1 = {};
#pragma unroll
        for (int k = 0; k < 8; k++) {
            v16h a = load_A(sW, 256, mt * 16, k * 32, lane);
            acc0 = wmma16(a, bf0[k], acc0);
            acc1 = wmma16(a, bf1[k], acc1);
        }
        int rb = mt * 16 + mb;
        v8h o0, o1;
#pragma unroll
        for (int r = 0; r < 8; r++) {
            float bb = out_b[rb + r];
            o0[r] = (_Float16)fmaxf(acc0[r] + bb, 0.f);
            o1[r] = (_Float16)fmaxf(acc1[r] + bb, 0.f);
        }
        *(v8h*)(act2 + (size_t)t * 256 + rb)        = o0;
        *(v8h*)(act2 + (size_t)(t + 16) * 256 + rb) = o1;
    }
}

// end head: out = end_w @ act2 + end_b -> f32 (B,256,L)
__global__ void __launch_bounds__(256) k_gemm_end(const _Float16* __restrict__ act2,
                                                  const _Float16* __restrict__ endW,
                                                  const float* __restrict__ end_b,
                                                  float* __restrict__ out) {
    __shared__ _Float16 sW[256 * 256];                 // 128 KB
    stage_lds_async(sW, endW, 256 * 256 * 2);
    __syncthreads();
    int wave = threadIdx.x >> 5, lane = threadIdx.x & 31;
    int tile = blockIdx.x * 8 + wave;
    if (tile >= NTP) return;
    int t0 = tile * 32;
    int t  = t0 + (lane & 15);
    int mb = (lane & 16) ? 8 : 0;

    v16h bf0[8], bf1[8];
#pragma unroll
    for (int k = 0; k < 8; k++) { bf0[k] = load_B(act2, 256, t0,      k * 32, lane);
                                  bf1[k] = load_B(act2, 256, t0 + 16, k * 32, lane); }

    int t1 = t + 16;
    bool live0 = (t < BL), live1 = (t1 < BL);
    int b0i = live0 ? (t / L_)  : 0, l0 = live0 ? (t - b0i * L_)  : 0;
    int b1i = live1 ? (t1 / L_) : 0, l1 = live1 ? (t1 - b1i * L_) : 0;

#pragma unroll
    for (int mt = 0; mt < 16; mt++) {
        v8f acc0 = {}, acc1 = {};
#pragma unroll
        for (int k = 0; k < 8; k++) {
            v16h a = load_A(sW, 256, mt * 16, k * 32, lane);
            acc0 = wmma16(a, bf0[k], acc0);
            acc1 = wmma16(a, bf1[k], acc1);
        }
        int rb = mt * 16 + mb;
#pragma unroll
        for (int r = 0; r < 8; r++) {
            int o = rb + r;
            if (live0) out[((size_t)b0i * 256 + o) * L_ + l0] = acc0[r] + end_b[o];
            if (live1) out[((size_t)b1i * 256 + o) * L_ + l1] = acc1[r] + end_b[o];
        }
    }
}

// ---------------- host ----------------

extern "C" void kernel_launch(void* const* d_in, const int* in_sizes, int n_in,
                              void* d_out, int out_size, void* d_ws, size_t ws_size,
                              hipStream_t stream) {
    (void)in_sizes; (void)n_in; (void)out_size; (void)ws_size;
    const float* mel      = (const float*)d_in[0];
    const float* wav      = (const float*)d_in[1];
    const float* up_w     = (const float*)d_in[2];
    const float* up_b     = (const float*)d_in[3];
    const float* causal_w = (const float*)d_in[4];
    const float* causal_b = (const float*)d_in[5];
    const float* cond_w   = (const float*)d_in[6];
    const float* cond_b   = (const float*)d_in[7];
    const float* dil_w    = (const float*)d_in[8];
    const float* dil_b    = (const float*)d_in[9];
    const float* skip_w   = (const float*)d_in[10];
    const float* skip_b   = (const float*)d_in[11];
    const float* res_w    = (const float*)d_in[12];
    const float* res_b    = (const float*)d_in[13];
    const float* out_w    = (const float*)d_in[14];
    const float* out_b    = (const float*)d_in[15];
    const float* end_w    = (const float*)d_in[16];
    const float* end_b    = (const float*)d_in[17];

    char* ws = (char*)d_ws;
    size_t off = 0;
    auto take = [&](size_t bytes) -> char* {
        char* p = ws + off;
        off += (bytes + 255) & ~(size_t)255;
        return p;
    };
    _Float16* mel_up = (_Float16*)take((size_t)BLP * 80 * 2);
    _Float16* hbuf   = (_Float16*)take((size_t)BLP * 128 * 2);
    _Float16* packed = (_Float16*)take((size_t)BLP * 320 * 2);
    _Float16* xg     = (_Float16*)take((size_t)BLP * 128 * 2);
    float*    skips  = (float*)   take((size_t)BLP * 256 * 4);
    _Float16* act2   = (_Float16*)take((size_t)BLP * 256 * 2);
    _Float16* W1     = (_Float16*)take((size_t)16 * 256 * 320 * 2);
    _Float16* A2     = (_Float16*)take((size_t)16 * 368 * 128 * 2);
    float*    gbias  = (float*)   take((size_t)16 * 256 * 4);
    float*    bias2  = (float*)   take((size_t)16 * 368 * 4);
    _Float16* outW   = (_Float16*)take((size_t)256 * 256 * 2);
    _Float16* endW   = (_Float16*)take((size_t)256 * 256 * 2);

    // weight prep
    k_prep_w1   <<<16 * 256 * 320 / 256, 256, 0, stream>>>(cond_w, dil_w, W1);
    k_prep_gbias<<<16, 256, 0, stream>>>(cond_b, dil_b, gbias);
    k_prep_a2   <<<16 * 368 * 128 / 256, 256, 0, stream>>>(skip_w, res_w, A2);
    k_prep_bias2<<<23, 256, 0, stream>>>(skip_b, res_b, bias2);
    k_prep_outw <<<256, 256, 0, stream>>>(out_w, outW);
    k_prep_endw <<<256, 256, 0, stream>>>(end_w, endW);

    // frontend
    k_upsample<<<(size_t)BLP * 80 / 256, 256, 0, stream>>>(mel, up_w, up_b, mel_up);
    k_causal  <<<(size_t)BLP * 128 / 256, 256, 0, stream>>>(wav, causal_w, causal_b, hbuf);
    k_zero    <<<2048, 256, 0, stream>>>(skips, (size_t)BLP * 256);

    // 16 residual blocks (sequential dependency via kernel boundaries)
    for (int i = 0; i < 16; i++) {
        int d = 1 << (i & 7);
        k_pack<<<(size_t)BLP * 320 / 256, 256, 0, stream>>>(mel_up, hbuf, packed, d);
        k_gemm_gate<<<NTP / 8, 256, 0, stream>>>(packed, W1 + (size_t)i * 256 * 320,
                                                 gbias + (size_t)i * 256, xg);
        k_gemm_sr<<<NTP / 8, 256, 0, stream>>>(xg, A2 + (size_t)i * 368 * 128,
                                               bias2 + (size_t)i * 368, skips, hbuf);
    }

    // heads
    k_gemm_out<<<NTP / 8, 256, 0, stream>>>(skips, outW, out_b, act2);
    k_gemm_end<<<NTP / 8, 256, 0, stream>>>(act2, endW, end_b, (float*)d_out);
}
